// CachedAttentionLayer_26723286515720
// MI455X (gfx1250) — compile-verified, matrix operand order checked by reading, not run on hardware
//
#include <hip/hip_runtime.h>
#include <hip/hip_bf16.h>
#include <math.h>

typedef __bf16 bf16;
typedef __attribute__((ext_vector_type(16))) __bf16 v16bf;
typedef __attribute__((ext_vector_type(8)))  __bf16 v8bf;
typedef __attribute__((ext_vector_type(4)))  __bf16 v4bf;
typedef __attribute__((ext_vector_type(8)))  float  v8f;
typedef __attribute__((ext_vector_type(4)))  float  v4f;

#define B_   2
#define T_   2048
#define D_   2048
#define H_   16
#define HKV_ 4
#define HD_  128
#define REP_ (H_ / HKV_)

__device__ __forceinline__ v8f v8f_zero() {
  v8f z;
#pragma unroll
  for (int i = 0; i < 8; ++i) z[i] = 0.0f;
  return z;
}

// Load one 16-bf16 A/B fragment for wave32 WMMA 16x16x32.
// ISA 7.12.2: lane L holds row/col (L%16); element e has
// K(e) = (e<8 ? e : e+8) + (L/16)*8  -> two contiguous 8-elem (16B) runs.
__device__ __forceinline__ v16bf load_frag_row(const bf16* rowbase, int lane) {
  const int half = (lane >> 4) & 1;
  v8bf lo = *(const v8bf*)(rowbase + half * 8);
  v8bf hi = *(const v8bf*)(rowbase + half * 8 + 16);
  v16bf r;
#pragma unroll
  for (int i = 0; i < 8; ++i) { r[i] = lo[i]; r[8 + i] = hi[i]; }
  return r;
}

__device__ __forceinline__ v8f wmma_bf16(v16bf a, v16bf b, v8f c) {
  return __builtin_amdgcn_wmma_f32_16x16x32_bf16(
      false, a, false, b, (short)0, c, false, false);
}

__device__ __forceinline__ bf16 cvt_bf16(float x) { return (bf16)x; }
__device__ __forceinline__ bf16 cvt_bf16(bf16 x)  { return x; }

// ----------------------------------------------------------------------------
// C[M,N] = cscale * (A[M,K] @ Bw[K,N]).  A is f32 (converted) or bf16;
// C is bf16 or f32.  Block: 256 threads (8 waves), tile 128x128, K-chunk 32.
// Software-pipelined: stage k+1 tile in registers while WMMAs consume tile k.
// cscale lets the caller fold e.g. the attention softmax scale into Q.
// ----------------------------------------------------------------------------
template <typename AT, typename CT>
__global__ __launch_bounds__(256) void gemm_wmma_kernel(
    const AT* __restrict__ A, const float* __restrict__ Bw, CT* __restrict__ C,
    int M, int N, int K, float cscale) {
  __shared__ __align__(16) bf16 As[128][40];  // [m][k], 80B rows (16B aligned)
  __shared__ __align__(16) bf16 Bs[128][40];  // [n][k] (transposed in LDS)

  const int tid  = threadIdx.x;
  const int lane = tid & 31;
  const int w    = tid >> 5;   // 0..7
  const int wm   = w >> 2;     // 0..1
  const int wn   = w & 3;      // 0..3
  const int m0   = blockIdx.y * 128;
  const int n0   = blockIdx.x * 128;
  const int col  = lane & 15;
  const int rbase = (lane >> 4) * 8;

  v8f acc[4][2];
#pragma unroll
  for (int mi = 0; mi < 4; ++mi)
#pragma unroll
    for (int ni = 0; ni < 2; ++ni) acc[mi][ni] = v8f_zero();

  // A-tile staging: thread -> row (t>>1), 16 consecutive k at (t&1)*16
  const int arow = tid >> 1;
  const int acol = (tid & 1) * 16;
  // B-tile staging: thread -> 4 k-rows x 4 n-cols sub-block (packable stores)
  const int k0   = (tid >> 5) * 4;   // 0..28
  const int nloc = (tid & 31) * 4;   // 0..124

  AT    ar[16];
  float br[4][4];

  auto load_stage = [&](int kk) {
    const AT* ag = A + (size_t)(m0 + arow) * K + kk + acol;
#pragma unroll
    for (int j = 0; j < 16; ++j) ar[j] = ag[j];
#pragma unroll
    for (int i = 0; i < 4; ++i) {
      v4f t = *(const v4f*)(Bw + (size_t)(kk + k0 + i) * N + n0 + nloc);
#pragma unroll
      for (int j = 0; j < 4; ++j) br[i][j] = t[j];
    }
  };

  auto store_stage = [&]() {
    v8bf p;
#pragma unroll
    for (int j = 0; j < 8; ++j) p[j] = cvt_bf16(ar[j]);
    *(v8bf*)&As[arow][acol] = p;
#pragma unroll
    for (int j = 0; j < 8; ++j) p[j] = cvt_bf16(ar[8 + j]);
    *(v8bf*)&As[arow][acol + 8] = p;
#pragma unroll
    for (int ni = 0; ni < 4; ++ni) {  // pack 4 k per n -> ds_store_b64
      v4bf pb;
#pragma unroll
      for (int i = 0; i < 4; ++i) pb[i] = (bf16)br[i][ni];
      *(v4bf*)&Bs[nloc + ni][k0] = pb;
    }
  };

  load_stage(0);
  for (int kk = 0; kk < K; kk += 32) {
    __syncthreads();   // previous compute's LDS reads complete
    store_stage();
    __syncthreads();
    if (kk + 32 < K) load_stage(kk + 32);  // overlap next loads with WMMAs

    v16bf afrag[4];
#pragma unroll
    for (int mi = 0; mi < 4; ++mi)
      afrag[mi] = load_frag_row(&As[wm * 64 + mi * 16 + col][0], lane);
#pragma unroll
    for (int ni = 0; ni < 2; ++ni) {
      v16bf bfrag = load_frag_row(&Bs[wn * 32 + ni * 16 + col][0], lane);
#pragma unroll
      for (int mi = 0; mi < 4; ++mi)
        acc[mi][ni] = wmma_bf16(afrag[mi], bfrag, acc[mi][ni]);
    }
  }

#pragma unroll
  for (int mi = 0; mi < 4; ++mi)
#pragma unroll
    for (int ni = 0; ni < 2; ++ni)
#pragma unroll
      for (int i = 0; i < 8; ++i) {
        const int m = m0 + wm * 64 + mi * 16 + rbase + i;
        const int n = n0 + wn * 32 + ni * 16 + col;
        C[(size_t)m * N + n] = (CT)(acc[mi][ni][i] * cscale);
      }
}

// ----------------------------------------------------------------------------
// Flash attention: one block = (b, h, 128 q-rows); 8 waves, 16 q-rows/wave.
// KV chunks of 32, software-pipelined global->reg->LDS staging.
// Q arrives pre-scaled by log2(e)/sqrt(HD), so scores feed exp2 directly.
// Row-sum l is carried as a 9th accumulator tile fed by P @ ones (one extra
// WMMA) and rides through the same alpha rescale as O; only the row-max
// needs a shuffle butterfly.
// ----------------------------------------------------------------------------
__global__ __launch_bounds__(256) void attn_wmma_kernel(
    const bf16* __restrict__ q, const bf16* __restrict__ k,
    const bf16* __restrict__ v, bf16* __restrict__ o) {
  __shared__ __align__(16) bf16 Kl[32][136];    // [kv][d]
  __shared__ __align__(16) bf16 Vl[128][40];    // [d][kv]   (transposed)
  __shared__ __align__(16) bf16 Sl[8][16][40];  // per-wave P scratch [row][kv]

  const int tid   = threadIdx.x;
  const int lane  = tid & 31;
  const int w     = tid >> 5;
  const int h     = blockIdx.y;
  const int b     = blockIdx.z;
  const int hkv   = h / REP_;
  const int q0    = blockIdx.x * 128 + w * 16;
  const int col   = lane & 15;
  const int rbase = (lane >> 4) * 8;

  // Resident Q fragments: 16 rows x 128 d -> 4 k-chunks of 32
  v16bf qfrag[4];
  {
    const bf16* qrow = q + (size_t)(b * T_ + q0 + col) * (H_ * HD_) + h * HD_;
#pragma unroll
    for (int kc = 0; kc < 4; ++kc) qfrag[kc] = load_frag_row(qrow + kc * 32, lane);
  }

  // All-ones B fragment: (P @ ones)[r][c] == rowsum(P[r])
  v16bf ones_frag;
#pragma unroll
  for (int i = 0; i < 16; ++i) ones_frag[i] = (bf16)1.0f;

  v8f o_acc[8];
#pragma unroll
  for (int nt = 0; nt < 8; ++nt) o_acc[nt] = v8f_zero();
  v8f l_acc = v8f_zero();   // running softmax denominator per owned row
  float m_i[8];
#pragma unroll
  for (int i = 0; i < 8; ++i) m_i[i] = -__builtin_inff();

  // K staging: thread -> kv row (t>>3), 16 d at (t&7)*16  (contiguous copy)
  const int jr = tid >> 3;
  const int db = (tid & 7) * 16;
  // V staging: thread -> 4 kv x 4 d sub-block, packed transposed b64 stores
  const int j0 = (tid >> 5) * 4;   // 0..28
  const int d0 = (tid & 31) * 4;   // 0..124

  v8bf kr0, kr1;
  v4bf vr[4];

  auto load_kv = [&](int kv0) {
    const bf16* kg = k + (size_t)(b * T_ + kv0 + jr) * (HKV_ * HD_) + hkv * HD_ + db;
    kr0 = *(const v8bf*)kg;
    kr1 = *(const v8bf*)(kg + 8);
    const bf16* vg = v + (size_t)(b * T_ + kv0 + j0) * (HKV_ * HD_) + hkv * HD_ + d0;
#pragma unroll
    for (int i = 0; i < 4; ++i) vr[i] = *(const v4bf*)(vg + (size_t)i * (HKV_ * HD_));
  };

  auto store_kv = [&]() {
    *(v8bf*)&Kl[jr][db] = kr0;
    *(v8bf*)&Kl[jr][db + 8] = kr1;
#pragma unroll
    for (int dd = 0; dd < 4; ++dd) {  // transpose in regs, pack 4 kv per d
      v4bf t;
#pragma unroll
      for (int i = 0; i < 4; ++i) t[i] = vr[i][dd];
      *(v4bf*)&Vl[d0 + dd][j0] = t;
    }
  };

  load_kv(0);
  for (int kv0 = 0; kv0 < T_; kv0 += 32) {
    __syncthreads();   // previous iteration's Kl/Vl/Sl reads complete
    store_kv();
    __syncthreads();
    if (kv0 + 32 < T_) load_kv(kv0 + 32);  // overlap with S/PV compute

    // S = Q K^T  (16 x 32): 2 n-tiles x 4 k-chunks (already exp2-scaled)
    v8f s0 = v8f_zero(), s1 = v8f_zero();
#pragma unroll
    for (int kc = 0; kc < 4; ++kc) {
      v16bf b0 = load_frag_row(&Kl[col][kc * 32], lane);
      v16bf b1 = load_frag_row(&Kl[16 + col][kc * 32], lane);
      s0 = wmma_bf16(qfrag[kc], b0, s0);
      s1 = wmma_bf16(qfrag[kc], b1, s1);
    }

    // Online softmax per owned row: only the row-max needs a butterfly
    // (shfl_xor masks 1..8 stay inside the 16-lane half that owns the row).
#pragma unroll
    for (int i = 0; i < 8; ++i) {
      float rmax = fmaxf(s0[i], s1[i]);
#pragma unroll
      for (int off = 1; off < 16; off <<= 1)
        rmax = fmaxf(rmax, __shfl_xor(rmax, off, 32));
      const float mn    = fmaxf(m_i[i], rmax);
      const float alpha = exp2f(m_i[i] - mn);
      m_i[i] = mn;
      const float p0 = exp2f(s0[i] - mn);
      const float p1 = exp2f(s1[i] - mn);
      l_acc[i] *= alpha;
#pragma unroll
      for (int nt = 0; nt < 8; ++nt) o_acc[nt][i] *= alpha;
      Sl[w][rbase + i][col]      = (bf16)p0;  // C-layout -> A-layout via LDS
      Sl[w][rbase + i][16 + col] = (bf16)p1;
    }
    __syncthreads();

    // O += P V (8 n-tiles) and l += P @ ones (9th WMMA; no shuffle reduction)
    v16bf pfrag = load_frag_row(&Sl[w][col][0], lane);
#pragma unroll
    for (int nt = 0; nt < 8; ++nt) {
      v16bf vfrag = load_frag_row(&Vl[nt * 16 + col][0], lane);
      o_acc[nt] = wmma_bf16(pfrag, vfrag, o_acc[nt]);
    }
    l_acc = wmma_bf16(pfrag, ones_frag, l_acc);
  }

  // Normalize and store attention output (bf16, [B*T, H*HD])
#pragma unroll
  for (int i = 0; i < 8; ++i) {
    const float inv = 1.0f / l_acc[i];
    const size_t row = (size_t)(b * T_ + q0 + rbase + i) * (H_ * HD_) + h * HD_;
#pragma unroll
    for (int nt = 0; nt < 8; ++nt)
      o[row + nt * 16 + col] = (bf16)(o_acc[nt][i] * inv);
  }
}

// ----------------------------------------------------------------------------
extern "C" void kernel_launch(void* const* d_in, const int* in_sizes, int n_in,
                              void* d_out, int out_size, void* d_ws, size_t ws_size,
                              hipStream_t stream) {
  (void)in_sizes; (void)n_in; (void)out_size; (void)ws_size;
  const float* x  = (const float*)d_in[0];
  const float* wq = (const float*)d_in[1];
  const float* wk = (const float*)d_in[2];
  const float* wv = (const float*)d_in[3];
  const float* wo = (const float*)d_in[4];
  float* out = (float*)d_out;

  // Workspace: bf16 Q, K, V, attn-out (~40 MB total)
  bf16* qb = (bf16*)d_ws;
  bf16* kb = qb + (size_t)B_ * T_ * H_ * HD_;
  bf16* vb = kb + (size_t)B_ * T_ * HKV_ * HD_;
  bf16* ab = vb + (size_t)B_ * T_ * HKV_ * HD_;

  const int M = B_ * T_;  // 4096
  dim3 blk(256);

  // log2(e)/sqrt(HD): folded into Q so attention scores feed exp2 directly
  const float q_scale = 0.12752259530291926f;

  // Projections (f32 in, bf16 out)
  gemm_wmma_kernel<float, bf16>
      <<<dim3((H_ * HD_) / 128, M / 128), blk, 0, stream>>>(x, wq, qb, M, H_ * HD_, D_, q_scale);
  gemm_wmma_kernel<float, bf16>
      <<<dim3((HKV_ * HD_) / 128, M / 128), blk, 0, stream>>>(x, wk, kb, M, HKV_ * HD_, D_, 1.0f);
  gemm_wmma_kernel<float, bf16>
      <<<dim3((HKV_ * HD_) / 128, M / 128), blk, 0, stream>>>(x, wv, vb, M, HKV_ * HD_, D_, 1.0f);

  // Flash attention (bf16 in/out)
  attn_wmma_kernel<<<dim3(T_ / 128, H_, B_), blk, 0, stream>>>(qb, kb, vb, ab);

  // Output projection (bf16 A, f32 weights/out)
  gemm_wmma_kernel<bf16, float>
      <<<dim3(D_ / 128, M / 128), blk, 0, stream>>>(ab, wo, out, M, D_, H_ * HD_, 1.0f);
}
